// SALAD_6571299962919
// MI455X (gfx1250) — matrix operand order repaired and checked by reading, not compiled, not run on hardware
//
#include <hip/hip_runtime.h>
#include <hip/hip_bf16.h>

// ---------------------------------------------------------------------------
// Types
// ---------------------------------------------------------------------------
typedef __bf16 bf16;
typedef __attribute__((ext_vector_type(16))) __bf16 bf16x16;
typedef __attribute__((ext_vector_type(8)))  float  f32x8;
typedef __attribute__((ext_vector_type(2)))  float  f32x2;
typedef __attribute__((ext_vector_type(4)))  int    v4i;

// pointer-to-int4 in explicit address spaces (per hipcc diagnostic for the
// async-to-LDS builtin: param0 is 'int __vector(4) __device__ *' = AS1 int4*)
typedef __attribute__((address_space(1))) v4i* as1_v4i_ptr;
typedef __attribute__((address_space(3))) v4i* as3_v4i_ptr;

union FragB16 {
    bf16x16 v;
    uint4   q[2];
};

static __device__ inline bf16 f2b(float f) { return (bf16)f; }

static __device__ inline unsigned pack2(float a, float b) {
    union { bf16 h[2]; unsigned u; } x;
    x.h[0] = (bf16)a; x.h[1] = (bf16)b;
    return x.u;
}

// ---- CDNA5 feature guards -------------------------------------------------
#if __has_builtin(__builtin_amdgcn_global_load_async_to_lds_b128)
#define HAS_ASYNC_LDS 1
#else
#define HAS_ASYNC_LDS 0
#endif

#if __has_builtin(__builtin_amdgcn_wmma_f32_16x16x4_f32)
#define HAS_WMMA_F32K4 1
#else
#define HAS_WMMA_F32K4 0
#endif

static __device__ inline void wait_async0() {
#if __has_builtin(__builtin_amdgcn_s_wait_asynccnt)
    __builtin_amdgcn_s_wait_asynccnt(0);
#else
    asm volatile("s_wait_asynccnt 0x0" ::: "memory");
#endif
}

#if HAS_ASYNC_LDS
static __device__ inline void async_copy16(const void* g, void* lds) {
    __builtin_amdgcn_global_load_async_to_lds_b128(
        (as1_v4i_ptr)(unsigned long long)g,
        (as3_v4i_ptr)(unsigned)(unsigned long long)lds, 0, 0);
}
#endif

// ---------------------------------------------------------------------------
// Problem constants
// ---------------------------------------------------------------------------
#define BATCH 64
#define CCH   1536
#define NPX   256      // H*W
#define MCL   64
#define LDIM  128
#define GDIM  256
#define HID   512

// ---------------------------------------------------------------------------
// Kernel 0: weight conversion / packing (fp32 -> bf16), stacking, t transpose
// ---------------------------------------------------------------------------
__global__ void prep_kernel(const float* __restrict__ Wc1, const float* __restrict__ bc1,
                            const float* __restrict__ Ws1, const float* __restrict__ bs1,
                            const float* __restrict__ Wc2, const float* __restrict__ Ws2,
                            const float* __restrict__ Wt1, const float* __restrict__ Wt2,
                            const float* __restrict__ t,
                            bf16* __restrict__ W1b, float* __restrict__ bias1,
                            bf16* __restrict__ Wc2b, bf16* __restrict__ Ws2b,
                            bf16* __restrict__ Wt1b, bf16* __restrict__ Wt2b,
                            bf16* __restrict__ tTb)
{
    int id = blockIdx.x * 256 + threadIdx.x;
    // stacked W1 = [Wc1 ; Ws1]  : [1024][1536]
    const int N1 = 1024 * CCH;
    if (id < N1) {
        int o = id / CCH, k = id % CCH;
        W1b[id] = f2b(o < HID ? Wc1[o * CCH + k] : Ws1[(o - HID) * CCH + k]);
        return;
    }
    id -= N1;
    if (id < 1024) { bias1[id] = id < HID ? bc1[id] : bs1[id - HID]; return; }
    id -= 1024;
    if (id < LDIM * HID) { Wc2b[id] = f2b(Wc2[id]); return; }
    id -= LDIM * HID;
    if (id < MCL * HID)  { Ws2b[id] = f2b(Ws2[id]); return; }
    id -= MCL * HID;
    if (id < HID * CCH)  { Wt1b[id] = f2b(Wt1[id]); return; }
    id -= HID * CCH;
    if (id < GDIM * HID) { Wt2b[id] = f2b(Wt2[id]); return; }
    id -= GDIM * HID;
    if (id < CCH * BATCH) {                 // tTb[k][b] = t[b][k]
        int k = id >> 6, bb = id & 63;
        tTb[id] = f2b(t[bb * CCH + k]);
        return;
    }
}

// ---------------------------------------------------------------------------
// Generic LDS-tiled bf16 WMMA GEMM:
//   Out[b][o][n] = act( sum_k W[o][k] * X[b][xRowOff+k][n] + bias[o] )
// Block = 128 threads (4 waves), tile 64(M) x 64(N), K stepped by 32.
// Wave w owns M rows [w*16, w*16+16), all 64 N columns (4 WMMA D tiles).
// Panels staged with GLOBAL_LOAD_ASYNC_TO_LDS_B128 when available.
// ---------------------------------------------------------------------------
template <bool XF32, bool RELU, bool OUTBF16>
__global__ void __launch_bounds__(128)
wmma_gemm(const bf16* __restrict__ W, const void* __restrict__ Xv,
          const float* __restrict__ bias, void* __restrict__ Out,
          int M, int K, int N, long xBatchStride, long outBatchStride, int xRowOff)
{
    alignas(16) __shared__ bf16 As[64 * 32];   // [64 rows of M][32 K]
    alignas(16) __shared__ bf16 Bs[32 * 64];   // [32 rows of K][64 N]

    const int b  = blockIdx.z;
    const int m0 = blockIdx.y * 64;
    const int n0 = blockIdx.x * 64;
    const int tid = threadIdx.x;
    const int w   = tid >> 5;
    const int L   = tid & 31;

    f32x8 acc[4] = {};

    for (int k0 = 0; k0 < K; k0 += 32) {
        // ---- stage A panel: W[m0..m0+63][k0..k0+31] -> As (row-major) ----
#pragma unroll
        for (int i = 0; i < 2; ++i) {
            int c   = tid + i * 128;           // 256 chunks of 8 bf16
            int row = c >> 2;
            int col = (c & 3) * 8;
#if HAS_ASYNC_LDS
            async_copy16(&W[(long)(m0 + row) * K + k0 + col], &As[row * 32 + col]);
#else
            *(uint4*)&As[row * 32 + col] =
                *(const uint4*)&W[(long)(m0 + row) * K + k0 + col];
#endif
        }
        if (k0 + 32 < K)
            __builtin_prefetch(&W[(long)(m0 + (tid >> 2)) * K + k0 + 32], 0, 0);

        // ---- stage B panel: X[b][k0..k0+31][n0..n0+63] -> Bs ----
        if constexpr (XF32) {
            const float* X = (const float*)Xv;
            int row = tid >> 2;                // 32 rows, 4 threads each
            int col = (tid & 3) * 16;
            const float* src =
                &X[b * xBatchStride + (long)(xRowOff + k0 + row) * N + n0 + col];
            uint4 q0, q1;
            q0.x = pack2(src[0],  src[1]);  q0.y = pack2(src[2],  src[3]);
            q0.z = pack2(src[4],  src[5]);  q0.w = pack2(src[6],  src[7]);
            q1.x = pack2(src[8],  src[9]);  q1.y = pack2(src[10], src[11]);
            q1.z = pack2(src[12], src[13]); q1.w = pack2(src[14], src[15]);
            *(uint4*)&Bs[row * 64 + col]     = q0;
            *(uint4*)&Bs[row * 64 + col + 8] = q1;
        } else {
            const bf16* X = (const bf16*)Xv;
#pragma unroll
            for (int i = 0; i < 2; ++i) {
                int c   = tid + i * 128;       // 256 chunks of 8 bf16
                int row = c >> 3;
                int col = (c & 7) * 8;
#if HAS_ASYNC_LDS
                async_copy16(&X[b * xBatchStride +
                                (long)(xRowOff + k0 + row) * N + n0 + col],
                             &Bs[row * 64 + col]);
#else
                *(uint4*)&Bs[row * 64 + col] =
                    *(const uint4*)&X[b * xBatchStride +
                                      (long)(xRowOff + k0 + row) * N + n0 + col];
#endif
            }
        }
#if HAS_ASYNC_LDS
        wait_async0();
#endif
        __syncthreads();

        // ---- A fragment (16x32 bf16, ISA layout) ----
        FragB16 aF;
        {
            int row = (w << 4) + (L & 15);
            int kh  = (L >> 4) * 8;            // lanes<16: K 0-7/16-23; else 8-15/24-31
            aF.q[0] = *(const uint4*)&As[row * 32 + kh];
            aF.q[1] = *(const uint4*)&As[row * 32 + 16 + kh];
        }
        // ---- B fragments (32x16 bf16) + WMMA ----
#pragma unroll
        for (int nt = 0; nt < 4; ++nt) {
            FragB16 bF;
#pragma unroll
            for (int s = 0; s < 16; ++s)
                bF.v[s] = Bs[(((L >> 4) * 8) + (s & 7) + ((s >> 3) * 16)) * 64 + nt * 16 + (L & 15)];
            acc[nt] = __builtin_amdgcn_wmma_f32_16x16x32_bf16(
                false, aF.v, false, bF.v, (short)0, acc[nt], false, false);
        }
        __syncthreads();
    }

    // ---- epilogue: D layout VGPR r -> M = r + 8*(lane>=16), N = lane&15 ----
    const int rbase = m0 + (w << 4) + ((L >> 4) << 3);
#pragma unroll
    for (int nt = 0; nt < 4; ++nt) {
        const int col = n0 + nt * 16 + (L & 15);
#pragma unroll
        for (int r = 0; r < 8; ++r) {
            const int row = rbase + r;
            float v = acc[nt][r] + bias[row];
            if constexpr (RELU) v = fmaxf(v, 0.0f);
            const long idx = b * outBatchStride + (long)row * N + col;
            if constexpr (OUTBF16) ((bf16*)Out)[idx]  = f2b(v);
            else                   ((float*)Out)[idx] = v;
        }
    }
}

// ---------------------------------------------------------------------------
// Sinkhorn (log-domain, 3 iters) with dustbin row.  One block per batch.
// S: [B][64][256] fp32, P: [B][64][256] fp32 (dustbin row dropped).
// ---------------------------------------------------------------------------
__global__ void __launch_bounds__(256)
sinkhorn_kernel(const float* __restrict__ S, const float* __restrict__ dustp,
                float* __restrict__ P)
{
    const int b = blockIdx.x;
    const int t = threadIdx.x;
    const int w = t >> 5, L = t & 31;
    const float dust = dustp[0];
    const float norm = -logf((float)(NPX + MCL));        // -log(320)
    const float logNmM = logf((float)(NPX - MCL));       //  log(192)

    __shared__ float u[MCL + 1];
    __shared__ float v[NPX];

    const float* Sb = S + (long)b * MCL * NPX;

    if (t < MCL + 1) u[t] = 0.0f;
    v[t] = 0.0f;
    __syncthreads();

    for (int it = 0; it < 3; ++it) {
        // u[r] = log_a[r] - lse_n(Ms[r][n] + v[n]); rows spread over 8 waves
        for (int r = w; r < MCL + 1; r += 8) {
            float vals[8];
            float mx = -1e30f;
#pragma unroll
            for (int j = 0; j < 8; ++j) {
                int n = L + j * 32;
                float x = (r < MCL ? Sb[r * NPX + n] : dust) + v[n];
                vals[j] = x;
                mx = fmaxf(mx, x);
            }
            for (int off = 16; off; off >>= 1) mx = fmaxf(mx, __shfl_xor(mx, off, 32));
            float sm = 0.0f;
#pragma unroll
            for (int j = 0; j < 8; ++j) sm += __expf(vals[j] - mx);
            for (int off = 16; off; off >>= 1) sm += __shfl_xor(sm, off, 32);
            if (L == 0)
                u[r] = (norm + (r == MCL ? logNmM : 0.0f)) - (mx + logf(sm));
        }
        __syncthreads();
        // v[n] = norm - lse_m(Ms[m][n] + u[m]); one column per thread
        {
            const int n = t;
            float mx = -1e30f;
            for (int m = 0; m < MCL + 1; ++m) {
                float x = (m < MCL ? Sb[m * NPX + n] : dust) + u[m];
                mx = fmaxf(mx, x);
            }
            float sm = 0.0f;
            for (int m = 0; m < MCL + 1; ++m) {
                float x = (m < MCL ? Sb[m * NPX + n] : dust) + u[m];
                sm += __expf(x - mx);
            }
            v[n] = norm - (mx + logf(sm));
        }
        __syncthreads();
    }

    // P = exp(Ms + u + v - norm), rows 0..63 only
    for (int idx = t; idx < MCL * NPX; idx += 256) {
        int m = idx >> 8, n = idx & 255;
        P[(long)b * MCL * NPX + idx] = __expf(Sb[idx] + u[m] + v[n] - norm);
    }
}

// ---------------------------------------------------------------------------
// vlad[b][l][m] = sum_n f[b][l][n] * p[b][m][n]
// Preferred: fp32 WMMA 16x16x4 (full fp32 accuracy at matrix-pipe rate).
// A frag: lane L holds f(M=L%16, K = 2*(L>=16)+{0,1})  -> one b64 load.
// B frag: lane L holds p(N=L%16, K = 2*(L>=16)+{0,1})  -> one b64 load.
// One block per batch, 4 waves; wave owns 32 l-rows x 64 m-cols.
// ---------------------------------------------------------------------------
#if HAS_WMMA_F32K4
__global__ void __launch_bounds__(128)
vlad_wmma_kernel(const float* __restrict__ F, const float* __restrict__ P,
                 float* __restrict__ V)
{
    const int b = blockIdx.x;
    const int w = threadIdx.x >> 5;
    const int L = threadIdx.x & 31;
    const float* Fb = F + (long)b * LDIM * NPX;
    const float* Pb = P + (long)b * MCL * NPX;

    f32x8 acc[2][4] = {};
    const int mrow = L & 15;
    const int koff = (L >> 4) * 2;

    for (int n0 = 0; n0 < NPX; n0 += 4) {
        f32x2 aF[2];
#pragma unroll
        for (int i = 0; i < 2; ++i) {
            const int l = (w * 2 + i) * 16 + mrow;
            aF[i] = *(const f32x2*)&Fb[l * NPX + n0 + koff];
        }
#pragma unroll
        for (int mt = 0; mt < 4; ++mt) {
            f32x2 bF = *(const f32x2*)&Pb[(mt * 16 + mrow) * NPX + n0 + koff];
#pragma unroll
            for (int i = 0; i < 2; ++i)
                acc[i][mt] = __builtin_amdgcn_wmma_f32_16x16x4_f32(
                    false, aF[i], false, bF, (short)0, acc[i][mt], false, false);
        }
    }
    // D: l = tile_l0 + r + 8*(L>=16), m = mt*16 + (L&15)
#pragma unroll
    for (int i = 0; i < 2; ++i) {
        const int lbase = (w * 2 + i) * 16 + ((L >> 4) << 3);
#pragma unroll
        for (int mt = 0; mt < 4; ++mt) {
            const int m = mt * 16 + (L & 15);
#pragma unroll
            for (int r = 0; r < 8; ++r)
                V[(long)b * LDIM * MCL + (lbase + r) * MCL + m] = acc[i][mt][r];
        }
    }
}
#endif

// fallback VALU vlad
__global__ void __launch_bounds__(256)
vlad_kernel(const float* __restrict__ F, const float* __restrict__ P,
            float* __restrict__ V)
{
    const int b = blockIdx.x;
    const int t = threadIdx.x;
    const float* Fb = F + (long)b * LDIM * NPX;
    const float* Pb = P + (long)b * MCL * NPX;
    for (int o = t; o < LDIM * MCL; o += 256) {
        const int l = o >> 6, m = o & 63;
        const float4* fr = (const float4*)&Fb[l * NPX];
        const float4* pr = (const float4*)&Pb[m * NPX];
        float s = 0.0f;
#pragma unroll 4
        for (int n4 = 0; n4 < NPX / 4; ++n4) {
            float4 a = fr[n4], c = pr[n4];
            s += a.x * c.x + a.y * c.y + a.z * c.z + a.w * c.w;
        }
        V[(long)b * LDIM * MCL + o] = s;
    }
}

// ---------------------------------------------------------------------------
// Final: intra-normalize vlad columns (over l), l2norm tok, concat,
// global l2norm.  One block per batch.
// out[b][0..255] = tok_n ; out[b][256 + l*64 + m] = vlad_n
// ---------------------------------------------------------------------------
__global__ void __launch_bounds__(256)
finalize_kernel(const float* __restrict__ V, const float* __restrict__ tokT,
                float* __restrict__ Out)
{
    const int b = blockIdx.x;
    const int t = threadIdx.x;
    __shared__ float cn[MCL];
    __shared__ float red[256];

    const float* Vb = V + (long)b * LDIM * MCL;
    float* outr = Out + (long)b * (GDIM + LDIM * MCL);

    // per-m column norms over l
    if (t < MCL) {
        float s = 0.0f;
        for (int l = 0; l < LDIM; ++l) { float x = Vb[l * MCL + t]; s += x * x; }
        cn[t] = fmaxf(sqrtf(s), 1e-12f);
    }
    // tok norm
    const float tokv = tokT[t * BATCH + b];
    red[t] = tokv * tokv;
    __syncthreads();
    for (int s2 = 128; s2; s2 >>= 1) {
        if (t < s2) red[t] += red[t + s2];
        __syncthreads();
    }
    const float tn = fmaxf(sqrtf(red[0]), 1e-12f);
    __syncthreads();

    // write normalized pieces, accumulate total squared
    float acc = 0.0f;
    {
        float v0 = tokv / tn;
        outr[t] = v0;
        acc += v0 * v0;
    }
    for (int o = t; o < LDIM * MCL; o += 256) {
        const int m = o & 63;
        float v0 = Vb[o] / cn[m];
        outr[GDIM + o] = v0;
        acc += v0 * v0;
    }
    red[t] = acc;
    __syncthreads();
    for (int s2 = 128; s2; s2 >>= 1) {
        if (t < s2) red[t] += red[t + s2];
        __syncthreads();
    }
    const float scale = 1.0f / fmaxf(sqrtf(red[0]), 1e-12f);
    __syncthreads();
    for (int i = t; i < GDIM + LDIM * MCL; i += 256) outr[i] *= scale;
}

// ---------------------------------------------------------------------------
// Host-side launcher
// ---------------------------------------------------------------------------
extern "C" void kernel_launch(void* const* d_in, const int* in_sizes, int n_in,
                              void* d_out, int out_size, void* d_ws, size_t ws_size,
                              hipStream_t stream)
{
    const float* x    = (const float*)d_in[0];
    const float* t    = (const float*)d_in[1];
    const float* Wt1  = (const float*)d_in[2];
    const float* bt1  = (const float*)d_in[3];
    const float* Wt2  = (const float*)d_in[4];
    const float* bt2  = (const float*)d_in[5];
    const float* Wc1  = (const float*)d_in[6];
    const float* bc1  = (const float*)d_in[7];
    const float* Wc2  = (const float*)d_in[8];
    const float* bc2  = (const float*)d_in[9];
    const float* Ws1  = (const float*)d_in[10];
    const float* bs1  = (const float*)d_in[11];
    const float* Ws2  = (const float*)d_in[12];
    const float* bs2  = (const float*)d_in[13];
    const float* dust = (const float*)d_in[14];

    char* ws = (char*)d_ws;
    // workspace layout (bytes, all 256-aligned)
    bf16*  W1b   = (bf16*)(ws + 0);                       // 1024*1536 bf16
    float* bias1 = (float*)(ws + 3145728);                // 1024 f32
    bf16*  Wc2b  = (bf16*)(ws + 3149824);                 // 128*512
    bf16*  Ws2b  = (bf16*)(ws + 3280896);                 // 64*512
    bf16*  Wt1b  = (bf16*)(ws + 3346432);                 // 512*1536
    bf16*  Wt2b  = (bf16*)(ws + 4919296);                 // 256*512
    bf16*  tTb   = (bf16*)(ws + 5181440);                 // 1536*64
    bf16*  H     = (bf16*)(ws + 5378048);                 // 64*1024*256
    bf16*  htb   = (bf16*)(ws + 38932480);                // 512*64
    float* fbuf  = (float*)(ws + 38998016);               // 64*128*256
    float* sbuf  = (float*)(ws + 47386624);               // 64*64*256
    float* tokT  = (float*)(ws + 51580928);               // 256*64
    float* pbuf  = (float*)(ws + 51646464);               // 64*64*256
    float* vladb = (float*)(ws + 55840768);               // 64*128*64

    // 0) pack/convert weights
    {
        const int total = 1024 * CCH + 1024 + LDIM * HID + MCL * HID +
                          HID * CCH + GDIM * HID + CCH * BATCH;
        prep_kernel<<<(total + 255) / 256, 256, 0, stream>>>(
            Wc1, bc1, Ws1, bs1, Wc2, Ws2, Wt1, Wt2, t,
            W1b, bias1, Wc2b, Ws2b, Wt1b, Wt2b, tTb);
    }

    // 1) layer-1 conv GEMM (fused Wc1/Ws1): H = relu(W1 @ xf + b1), bf16 out
    wmma_gemm<true, true, true><<<dim3(NPX / 64, 1024 / 64, BATCH), 128, 0, stream>>>(
        W1b, (const void*)x, bias1, (void*)H,
        1024, CCH, NPX, (long)CCH * NPX, (long)1024 * NPX, 0);

    // 1b) tok layer 1: htb = relu(Wt1 @ t^T + bt1), bf16 out, N = 64
    wmma_gemm<false, true, true><<<dim3(1, HID / 64, 1), 128, 0, stream>>>(
        Wt1b, (const void*)tTb, bt1, (void*)htb,
        HID, CCH, BATCH, 0L, 0L, 0);

    // 2) layer-2: f = Wc2 @ H[0:512] + bc2   (f32 out)
    wmma_gemm<false, false, false><<<dim3(NPX / 64, LDIM / 64, BATCH), 128, 0, stream>>>(
        Wc2b, (const void*)H, bc2, (void*)fbuf,
        LDIM, HID, NPX, (long)1024 * NPX, (long)LDIM * NPX, 0);

    // 2b) layer-2: s = Ws2 @ H[512:1024] + bs2 (f32 out)
    wmma_gemm<false, false, false><<<dim3(NPX / 64, MCL / 64, BATCH), 128, 0, stream>>>(
        Ws2b, (const void*)H, bs2, (void*)sbuf,
        MCL, HID, NPX, (long)1024 * NPX, (long)MCL * NPX, HID);

    // 2c) tok layer 2: tokT = Wt2 @ htb + bt2 (f32 out, [256][64])
    wmma_gemm<false, false, false><<<dim3(1, GDIM / 64, 1), 128, 0, stream>>>(
        Wt2b, (const void*)htb, bt2, (void*)tokT,
        GDIM, HID, BATCH, 0L, 0L, 0);

    // 3) Sinkhorn assignment probs
    sinkhorn_kernel<<<BATCH, 256, 0, stream>>>(sbuf, dust, pbuf);

    // 4) vlad = f @ p^T  (fp32 WMMA when available)
#if HAS_WMMA_F32K4
    vlad_wmma_kernel<<<BATCH, 128, 0, stream>>>(fbuf, pbuf, vladb);
#else
    vlad_kernel<<<BATCH, 256, 0, stream>>>(fbuf, pbuf, vladb);
#endif

    // 5) normalize + concat + global l2norm
    finalize_kernel<<<BATCH, 256, 0, stream>>>(vladb, tokT, (float*)d_out);
}
